// FlowAttention_10144712753256
// MI455X (gfx1250) — compile-verified, matrix operand order checked by reading, not run on hardware
//
#include <hip/hip_runtime.h>
#include <math.h>

// ---------------------------------------------------------------------------
// FlowAttention for MI455X (gfx1250).  B=8, L=S=4096, H=8, D=64.
// Memory-bound: stage sigmoid(q)/sigmoid(k) once as f16; run both per-head
// GEMMs through v_wmma_f32_16x16x32_f16 (f32 accumulate); V tiles staged via
// GLOBAL_LOAD_ASYNC_TO_LDS_B128 (ASYNCcnt) when the toolchain exposes it.
// ---------------------------------------------------------------------------

typedef __attribute__((ext_vector_type(16))) _Float16 v16h;
typedef __attribute__((ext_vector_type(8)))  float    v8f;
typedef __attribute__((ext_vector_type(4)))  int      v4i;

#define B_  8
#define H_  8
#define L_  4096
#define D_  64
#define BH_ (B_ * H_)
#define KSPLIT_ 8          // K-dimension split of the kv GEMM
#define EPS 1e-6f
// element offset of (b,l,h,d) in the [B,L,H,D] tensors
#define GIDX(b, l, h, d) ((((size_t)(b) * L_ + (l)) * H_ + (h)) * D_ + (d))

__device__ __forceinline__ float sigmoidf_(float x) {
  return 1.0f / (1.0f + __expf(-x));
}

// --- CDNA5 async global->LDS copy (b128) with compile-safe fallback --------
// Probe-learned signature: (v4i addrspace(1)*, v4i addrspace(3)*, Imm, Imm)
__device__ __forceinline__ void async_copy_b128(const float* g, float* l) {
#if __has_builtin(__builtin_amdgcn_global_load_async_to_lds_b128)
  __builtin_amdgcn_global_load_async_to_lds_b128(
      (__attribute__((address_space(1))) v4i*)const_cast<float*>(g),
      (__attribute__((address_space(3))) v4i*)l, 0, 0);
#else
  *(float4*)l = *(const float4*)g;
#endif
}

__device__ __forceinline__ void wait_async0() {
#if __has_builtin(__builtin_amdgcn_s_wait_asynccnt)
  __builtin_amdgcn_s_wait_asynccnt(0);
#else
  asm volatile("s_wait_asynccnt 0" ::: "memory");
#endif
}

// ---------------------------------------------------------------- kernel 0 --
// zero the four [BH,64] reduction buffers + the f32 kv accumulator
// (4*BH*64 + BH*64*64 = 278,528 floats, laid out contiguously)
__global__ void fa_zero(float* p, int n) {
  int i = blockIdx.x * blockDim.x + threadIdx.x;
  if (i < n) p[i] = 0.0f;
}

// ---------------------------------------------------------------- kernel 1 --
// qs = sigmoid(q), ks = sigmoid(k) staged as f16 in [BH][L][D];
// q_sum/k_sum[bh][d] = column sums over L (atomic f32 reduction).
// grid (BH, 16) x 256 threads; each block handles 256 rows of one head.
__global__ void fa_sigmoid_sums(const float* __restrict__ q,
                                const float* __restrict__ k,
                                _Float16* __restrict__ qsH,
                                _Float16* __restrict__ ksH,
                                float* __restrict__ qSum,
                                float* __restrict__ kSum) {
  const int bh = blockIdx.x, b = bh >> 3, h = bh & 7;
  const int l0 = blockIdx.y * 256;
  const int t = threadIdx.x, d = t & 63, rg = t >> 6;  // 4 row-groups
  float sq = 0.f, sk = 0.f;
  for (int i = 0; i < 64; ++i) {
    const int l = l0 + rg + 4 * i;
    const size_t gi = GIDX(b, l, h, d);
    const float qv = sigmoidf_(q[gi]);
    const float kv = sigmoidf_(k[gi]);
    const size_t wi = ((size_t)bh * L_ + l) * D_ + d;
    qsH[wi] = (_Float16)qv;
    ksH[wi] = (_Float16)kv;
    sq += qv;
    sk += kv;
  }
  __shared__ float rq[256], rk[256];
  rq[t] = sq;
  rk[t] = sk;
  __syncthreads();
  if (rg == 0) {
    const float aq = rq[d] + rq[d + 64] + rq[d + 128] + rq[d + 192];
    const float ak = rk[d] + rk[d + 64] + rk[d + 128] + rk[d + 192];
    atomicAdd(&qSum[bh * D_ + d], aq);
    atomicAdd(&kSum[bh * D_ + d], ak);
  }
}

// ---------------------------------------------------------------- kernel 2 --
// nRow[l] = 1/dot(qs[l]+EPS, kSum+EPS);  nCol[s] = 1/dot(ks[s]+EPS, qSum+EPS)
// qnSum[d] += sum_l qs[l,d]*nRow[l];     knSum[d] += sum_s ks[s,d]*nCol[s]
// grid (BH, 16) x 256 threads, one row per thread.
__global__ void fa_normalizers(const _Float16* __restrict__ qsH,
                               const _Float16* __restrict__ ksH,
                               const float* __restrict__ qSum,
                               const float* __restrict__ kSum,
                               float* __restrict__ nRow,
                               float* __restrict__ nCol,
                               float* __restrict__ qnSum,
                               float* __restrict__ knSum) {
  const int bh = blockIdx.x;
  const int l = blockIdx.y * 256 + threadIdx.x;
  const int t = threadIdx.x;
  __shared__ float sQ[D_], sK[D_], qnl[D_], knl[D_];
  if (t < D_) {
    sQ[t] = qSum[bh * D_ + t];
    sK[t] = kSum[bh * D_ + t];
    qnl[t] = 0.f;
    knl[t] = 0.f;
  }
  __syncthreads();

  const _Float16* qrow = qsH + ((size_t)bh * L_ + l) * D_;
  const _Float16* krow = ksH + ((size_t)bh * L_ + l) * D_;
  float dq = 0.f, dk = 0.f;
  for (int d = 0; d < D_; ++d) {
    dq += ((float)qrow[d] + EPS) * (sK[d] + EPS);
    dk += ((float)krow[d] + EPS) * (sQ[d] + EPS);
  }
  const float nr = 1.0f / dq;
  const float nc = 1.0f / dk;
  nRow[bh * L_ + l] = nr;
  nCol[bh * L_ + l] = nc;
  for (int d = 0; d < D_; ++d) {
    atomicAdd(&qnl[d], (float)qrow[d] * nr);
    atomicAdd(&knl[d], (float)krow[d] * nc);
  }
  __syncthreads();
  if (t < D_) {
    atomicAdd(&qnSum[bh * D_ + t], qnl[t]);
    atomicAdd(&knSum[bh * D_ + t], knl[t]);
  }
}

// ---------------------------------------------------------------- kernel 3 --
// nrRef[l] = sigmoid(dot(qs[l]+EPS, knSum+EPS) * (L/S))
// ncRef[s] = softmax_s(dot(ks[s]+EPS, qnSum+EPS)) * S
// grid BH x 512 threads; one block per head -> softmax needs no cross-block
// reduction. Each thread owns 8 rows.
__global__ void fa_refine(const _Float16* __restrict__ qsH,
                          const _Float16* __restrict__ ksH,
                          const float* __restrict__ qnSum,
                          const float* __restrict__ knSum,
                          float* __restrict__ nrRef,
                          float* __restrict__ ncRef) {
  const int bh = blockIdx.x;
  const int t = threadIdx.x;
  __shared__ float sQn[D_], sKn[D_], red[512];
  if (t < D_) {
    sQn[t] = qnSum[bh * D_ + t];
    sKn[t] = knSum[bh * D_ + t];
  }
  __syncthreads();

  // q side (pure elementwise; L/S == 1)
  for (int r = 0; r < 8; ++r) {
    const int l = r * 512 + t;
    const _Float16* qrow = qsH + ((size_t)bh * L_ + l) * D_;
    float dot = 0.f;
    for (int d = 0; d < D_; ++d) dot += ((float)qrow[d] + EPS) * (sKn[d] + EPS);
    nrRef[bh * L_ + l] = sigmoidf_(dot);
  }

  // k side: logits then block softmax
  float lg[8];
  float mx = -3.4e38f;
  for (int r = 0; r < 8; ++r) {
    const int s = r * 512 + t;
    const _Float16* krow = ksH + ((size_t)bh * L_ + s) * D_;
    float dot = 0.f;
    for (int d = 0; d < D_; ++d) dot += ((float)krow[d] + EPS) * (sQn[d] + EPS);
    lg[r] = dot;
    mx = fmaxf(mx, dot);
  }
  red[t] = mx;
  __syncthreads();
  for (int off = 256; off > 0; off >>= 1) {
    if (t < off) red[t] = fmaxf(red[t], red[t + off]);
    __syncthreads();
  }
  const float gmax = red[0];
  __syncthreads();
  float lsum = 0.f;
  for (int r = 0; r < 8; ++r) {
    lg[r] = __expf(lg[r] - gmax);
    lsum += lg[r];
  }
  red[t] = lsum;
  __syncthreads();
  for (int off = 256; off > 0; off >>= 1) {
    if (t < off) red[t] += red[t + off];
    __syncthreads();
  }
  const float inv = (float)L_ / red[0];
  for (int r = 0; r < 8; ++r) ncRef[bh * L_ + r * 512 + t] = lg[r] * inv;
}

// ---------------------------------------------------------------- kernel 4 --
// kv[d,e] = sum_s (ks[s,d]*ncRef[s]) * v[s,e];  64x64 per head, K=4096.
// grid (BH, KSPLIT) x 512 (16 waves = the 16 C tiles); each block reduces a
// 512-row K slab (16 chunks of 32) and atomically accumulates its partial
// f32 tile. The raw V tile (8KB) is staged with async global->LDS b128 copies
// (one per thread); the scaled-K tile is computed into LDS as f16.
__global__ void fa_kv_wmma(const _Float16* __restrict__ ksH,
                           const float* __restrict__ v,
                           const float* __restrict__ ncRef,
                           float* __restrict__ kvF) {
  const int bh = blockIdx.x, b = bh >> 3, h = bh & 7;
  const int t = threadIdx.x, lane = t & 31, wid = t >> 5;
  const int tileM = wid >> 2, tileN = wid & 3;  // 4x4 tile grid of kv
  const int m = lane & 15;                      // M row / N col for this lane
  const int kb = (lane >> 4) * 8;               // K sub-offset (A/B layouts)
  __shared__ _Float16 aT[32 * 64];  // ks*nc chunk [s_local][d]  (4 KB)
  __shared__ float    bTf[32 * 64]; // raw v chunk [s_local][e]  (8 KB)

  const int sBase = blockIdx.y * (L_ / KSPLIT_);
  v8f acc = {};
  for (int c = 0; c < (L_ / KSPLIT_) / 32; ++c) {
    const int s0 = sBase + c * 32;
    // async-stage raw v rows: 32x64 f32 = 8KB = 512 x b128 (one per thread)
    {
      const int sl = t >> 4, e0 = (t & 15) * 4;
      async_copy_b128(&v[GIDX(b, s0 + sl, h, e0)], &bTf[sl * 64 + e0]);
    }
    // manual stage of ks * ncRef as f16 (4 halves per thread)
    for (int j = 0; j < 4; ++j) {
      const int idx = t * 4 + j;
      const int sl = idx >> 6, d = idx & 63;
      const float sc = ncRef[bh * L_ + s0 + sl];
      aT[idx] = (_Float16)((float)ksH[((size_t)bh * L_ + s0 + sl) * D_ + d] * sc);
    }
    wait_async0();
    __syncthreads();
    v16h af, bf;
    for (int tt = 0; tt < 16; ++tt) {
      const int kl = (tt < 8) ? (kb + tt) : (16 + kb + (tt - 8));
      af[tt] = aT[kl * 64 + tileM * 16 + m];              // A[m, kl]
      bf[tt] = (_Float16)bTf[kl * 64 + tileN * 16 + m];   // B[kl, n]
    }
    acc = __builtin_amdgcn_wmma_f32_16x16x32_f16(false, af, false, bf,
                                                 (short)0, acc, false, false);
    __syncthreads();
  }
  // C layout: VGPR r -> (M = r + 8*(lane>=16), N = lane&15)
  for (int r = 0; r < 8; ++r) {
    const int mm = r + ((lane >> 4) << 3);
    atomicAdd(&kvF[(size_t)bh * D_ * D_ + (tileM * 16 + mm) * D_ +
                   tileN * 16 + (lane & 15)],
              acc[r]);
  }
}

// ---------------------------------------------------------------- kernel 5 --
// x[l,e] = (qs[l,:] @ kv) * nRow[l] * nrRef[l], written back to [B,L,H,D].
// grid (BH, 32) x 256 (8 waves); each block: 128 rows, kv (f32->f16) in LDS.
__global__ void fa_out_wmma(const _Float16* __restrict__ qsH,
                            const float* __restrict__ kvF,
                            const float* __restrict__ nRow,
                            const float* __restrict__ nrRef,
                            float* __restrict__ out) {
  const int bh = blockIdx.x, b = bh >> 3, h = bh & 7;
  const int t = threadIdx.x, lane = t & 31, wid = t >> 5;
  __shared__ _Float16 kvT[D_ * D_];  // 8 KB
  for (int j = 0; j < 16; ++j)
    kvT[j * 256 + t] = (_Float16)kvF[(size_t)bh * D_ * D_ + j * 256 + t];
  __syncthreads();

  const int lBase = blockIdx.y * 128 + wid * 16;  // wave's 16 output rows
  const int m = lane & 15;
  const int kb = (lane >> 4) * 8;

  // A fragments for both K chunks (shared by all 4 N tiles)
  v16h af0, af1;
  const _Float16* qrow = qsH + ((size_t)bh * L_ + lBase + m) * D_;
  for (int tt = 0; tt < 16; ++tt) {
    const int kl = (tt < 8) ? (kb + tt) : (16 + kb + (tt - 8));
    af0[tt] = qrow[kl];
    af1[tt] = qrow[32 + kl];
  }
  for (int nTile = 0; nTile < 4; ++nTile) {
    v8f acc = {};
    v16h bf;
    for (int tt = 0; tt < 16; ++tt) {
      const int kl = (tt < 8) ? (kb + tt) : (16 + kb + (tt - 8));
      bf[tt] = kvT[kl * 64 + nTile * 16 + m];
    }
    acc = __builtin_amdgcn_wmma_f32_16x16x32_f16(false, af0, false, bf,
                                                 (short)0, acc, false, false);
    for (int tt = 0; tt < 16; ++tt) {
      const int kl = 32 + ((tt < 8) ? (kb + tt) : (16 + kb + (tt - 8)));
      bf[tt] = kvT[kl * 64 + nTile * 16 + m];
    }
    acc = __builtin_amdgcn_wmma_f32_16x16x32_f16(false, af1, false, bf,
                                                 (short)0, acc, false, false);
    for (int r = 0; r < 8; ++r) {
      const int mm = r + ((lane >> 4) << 3);
      const int l = lBase + mm;
      const float sc = nRow[bh * L_ + l] * nrRef[bh * L_ + l];
      out[GIDX(b, l, h, nTile * 16 + (lane & 15))] = acc[r] * sc;
    }
  }
}

// ---------------------------------------------------------------------------
extern "C" void kernel_launch(void* const* d_in, const int* in_sizes, int n_in,
                              void* d_out, int out_size, void* d_ws, size_t ws_size,
                              hipStream_t stream) {
  const float* q = (const float*)d_in[0];
  const float* k = (const float*)d_in[1];
  const float* v = (const float*)d_in[2];
  float* out = (float*)d_out;

  // workspace layout
  const size_t nQK = (size_t)BH_ * L_ * D_;  // 16,777,216 halves each
  _Float16* qsH = (_Float16*)d_ws;
  _Float16* ksH = qsH + nQK;
  float* fb = (float*)(ksH + nQK);
  // contiguous zero-init region: 4 sum buffers + f32 kv accumulator
  float* qSum = fb;                          // [BH,64]
  float* kSum = qSum + BH_ * D_;
  float* qnSum = kSum + BH_ * D_;
  float* knSum = qnSum + BH_ * D_;
  float* kvF = knSum + BH_ * D_;             // [BH,64,64] f32
  float* nRow = kvF + (size_t)BH_ * D_ * D_; // [BH,L] x4
  float* nCol = nRow + (size_t)BH_ * L_;
  float* nrRef = nCol + (size_t)BH_ * L_;
  float* ncRef = nrRef + (size_t)BH_ * L_;

  const int nZero = 4 * BH_ * D_ + BH_ * D_ * D_;
  fa_zero<<<(nZero + 255) / 256, 256, 0, stream>>>(qSum, nZero);
  fa_sigmoid_sums<<<dim3(BH_, 16), 256, 0, stream>>>(q, k, qsH, ksH, qSum, kSum);
  fa_normalizers<<<dim3(BH_, 16), 256, 0, stream>>>(qsH, ksH, qSum, kSum,
                                                    nRow, nCol, qnSum, knSum);
  fa_refine<<<BH_, 512, 0, stream>>>(qsH, ksH, qnSum, knSum, nrRef, ncRef);
  fa_kv_wmma<<<dim3(BH_, KSPLIT_), 512, 0, stream>>>(ksH, v, ncRef, kvF);
  fa_out_wmma<<<dim3(BH_, 32), 256, 0, stream>>>(qsH, kvF, nRow, nrRef, out);
}
